// CustomMultiheadAttention_80315888435530
// MI455X (gfx1250) — matrix operand order repaired and checked by reading, not compiled, not run on hardware
//
#include <hip/hip_runtime.h>
#include <hip/hip_bf16.h>

// ---------------------------------------------------------------------------
// CDNA5 (gfx1250, wave32) fused multi-head attention
//   proj(Q/K/V) -> flash attention -> proj(O)
//   - all matmuls on v_wmma_f32_16x16x32_bf16 (fp32 -> bf16 operands, f32 acc)
//   - K/V tiles staged to LDS with global_load_async_to_lds_b128 (ASYNCcnt),
//     double-buffered so DMA overlaps WMMA/softmax
//   - row-sum of softmax numerator computed by an extra WMMA against a
//     ones matrix (replaces half of the ds_bpermute reductions)
// ---------------------------------------------------------------------------

typedef __bf16 bf16;
typedef __attribute__((ext_vector_type(16))) __bf16        v16bf;
typedef __attribute__((ext_vector_type(8)))  float         v8f;
typedef __attribute__((ext_vector_type(4)))  unsigned int  u32x4;

#define EMBED  1024
#define HEADS  16
#define HDIM   64
#define BATCH  2
#define SEQ    2048

// ---------------------------------------------------------------------------
// WMMA wrapper: D = A(16x32 bf16) * B(32x16 bf16) + C(16x16 f32)
// ---------------------------------------------------------------------------
static __device__ inline v8f wmma_bf16(v16bf a, v16bf b, v8f c) {
  return __builtin_amdgcn_wmma_f32_16x16x32_bf16(
      /*neg_a=*/false, a, /*neg_b=*/false, b,
      /*c_mod=*/(short)0, c, /*reuse_a=*/false, /*reuse_b=*/false);
}

union BfPack { u32x4 u[2]; v16bf v; };

// A operand, 16x32 bf16 (row m = lane%16).
// Lanes 0-15 hold K=0..15 contiguous, lanes 16-31 hold K=16..31.
static __device__ inline v16bf ld_a_bf16(const bf16* row, int hi) {
  const char* p = (const char*)row + hi * 32;
  BfPack t;
  t.u[0] = *(const u32x4*)(p);
  t.u[1] = *(const u32x4*)(p + 16);
  return t.v;
}

// B operand, 32x16 bf16 (col n = lane%16; row pointer is the column's source
// row).  Lanes 0-15 hold K=0..15 contiguous, lanes 16-31 hold K=16..31.
static __device__ inline v16bf ld_b_bf16(const bf16* row, int hi) {
  const char* p = (const char*)row + hi * 32;
  BfPack t;
  t.u[0] = *(const u32x4*)(p);
  t.u[1] = *(const u32x4*)(p + 16);
  return t.v;
}

// fp32-sourced operand loads with on-the-fly bf16 conversion.
static __device__ inline v16bf ld_a_f32(const float* row, int hi) {
  const float* p = row + hi * 16;
  v16bf r;
#pragma unroll
  for (int j = 0; j < 8; ++j) r[j] = (bf16)p[j];
#pragma unroll
  for (int j = 0; j < 8; ++j) r[j + 8] = (bf16)p[j + 8];
  return r;
}
static __device__ inline v16bf ld_b_f32(const float* row, int hi) {
  const float* p = row + hi * 16;
  v16bf r;
#pragma unroll
  for (int j = 0; j < 16; ++j) r[j] = (bf16)p[j];
  return r;
}

// ---------------------------------------------------------------------------
// Async global -> LDS helpers (CDNA5 ASYNCcnt path).
// Low 32 bits of a generic pointer into the shared segment are the LDS byte
// offset (ISA: LDS aperture check uses addr[63:32]; the LDS address is the low 32 bits).
// ---------------------------------------------------------------------------
static __device__ inline unsigned lds_off(const void* p) {
  return (unsigned)(unsigned long long)(size_t)p;
}
static __device__ inline void async_ld16(unsigned lds_byte, const void* g) {
  asm volatile("global_load_async_to_lds_b128 %0, %1, off"
               :
               : "v"(lds_byte), "v"((unsigned long long)(size_t)g)
               : "memory");
}
static __device__ inline void wait_asynccnt0() {
  asm volatile("s_wait_asynccnt 0x0" ::: "memory");
}

// ---------------------------------------------------------------------------
// GEMM: C[M,N] = A[M,K] * W[N,K]^T + bias   (nn.Linear semantics)
//   OUT_MODE 0: f32, plain [M,N]            (final output)
//   OUT_MODE 1: bf16, head-split  [B,H,S,D] (Q, K)
//   OUT_MODE 2: bf16, head-split+T [B,H,D,S](V)
// Block: 256 thr = 8 waves, tile 128(M) x 64(N); wave tile 32x32.
// ---------------------------------------------------------------------------
template <bool A_BF16, int OUT_MODE>
__global__ void __launch_bounds__(256)
gemm_wmma_kernel(const void* __restrict__ Aptr, const float* __restrict__ W,
                 const float* __restrict__ bias, void* __restrict__ out,
                 int M, int N, int K) {
  const int lane = threadIdx.x & 31;
  const int wave = threadIdx.x >> 5;
  const int hi   = lane >> 4;
  const int ln   = lane & 15;

  const int m0 = blockIdx.x * 128 + (wave & 3) * 32;
  const int n0 = blockIdx.y * 64  + (wave >> 2) * 32;

  v8f c[2][2] = {};

  for (int kk = 0; kk < K; kk += 32) {
    v16bf a0, a1;
    if (A_BF16) {
      const bf16* A = (const bf16*)Aptr;
      a0 = ld_a_bf16(A + (size_t)(m0 + ln) * K + kk, hi);
      a1 = ld_a_bf16(A + (size_t)(m0 + 16 + ln) * K + kk, hi);
    } else {
      const float* A = (const float*)Aptr;
      a0 = ld_a_f32(A + (size_t)(m0 + ln) * K + kk, hi);
      a1 = ld_a_f32(A + (size_t)(m0 + 16 + ln) * K + kk, hi);
    }
    v16bf b0 = ld_b_f32(W + (size_t)(n0 + ln) * K + kk, hi);
    v16bf b1 = ld_b_f32(W + (size_t)(n0 + 16 + ln) * K + kk, hi);

    c[0][0] = wmma_bf16(a0, b0, c[0][0]);
    c[0][1] = wmma_bf16(a0, b1, c[0][1]);
    c[1][0] = wmma_bf16(a1, b0, c[1][0]);
    c[1][1] = wmma_bf16(a1, b1, c[1][1]);
  }

  // C/D layout: lane holds column n = ln; VGPR r holds row m = r + 8*hi.
#pragma unroll
  for (int nt = 0; nt < 2; ++nt) {
    const int n = n0 + nt * 16 + ln;
    const float bv = bias[n];
#pragma unroll
    for (int mt = 0; mt < 2; ++mt) {
#pragma unroll
      for (int r = 0; r < 8; ++r) {
        const int m = m0 + mt * 16 + r + hi * 8;
        const float val = c[mt][nt][r] + bv;
        if constexpr (OUT_MODE == 0) {
          ((float*)out)[(size_t)m * N + n] = val;
        } else {
          const int bb = m / SEQ, s = m % SEQ;
          const int h = n >> 6, d = n & 63;
          if constexpr (OUT_MODE == 1) {
            ((bf16*)out)[(((size_t)bb * HEADS + h) * SEQ + s) * HDIM + d] =
                (bf16)val;
          } else {
            ((bf16*)out)[(((size_t)bb * HEADS + h) * HDIM + d) * SEQ + s] =
                (bf16)val;
          }
        }
      }
    }
  }
}

// ---------------------------------------------------------------------------
// Flash attention.  Grid (S/64, H, B), block 128 = 4 waves.
// Each wave owns 16 query rows; the block cooperatively streams 32-key K/V
// tiles through double-buffered LDS using async global->LDS DMA.
//   Q  : [B,H,S,D] bf16   (A operand, D-contiguous, held in registers)
//   K  : [B,H,S,D] bf16   -> Kl tile [32 keys][64 d]
//   Vt : [B,H,D,S] bf16   -> Vl tile [64 d][32 keys]
//   AO : [B,S,E]   bf16   (heads re-merged, feeds output projection)
// ---------------------------------------------------------------------------

// Stage one 32-key K/V tile into LDS: 8 KB total, 4 async b128 per thread.
static __device__ inline void stage_kv(unsigned klds, unsigned vlds,
                                       const bf16* Kh, const bf16* Vh,
                                       int kb, int tid) {
#pragma unroll
  for (int it = 0; it < 2; ++it) {
    const int c   = tid + it * 128;        // 256 x 16B chunks for K (32x64)
    const int row = c >> 3;                // key row
    const int off = (c & 7) * 8;           // element offset within row
    async_ld16(klds + (unsigned)(row * HDIM + off) * 2,
               Kh + (size_t)(kb + row) * HDIM + off);
  }
#pragma unroll
  for (int it = 0; it < 2; ++it) {
    const int c   = tid + it * 128;        // 256 x 16B chunks for V (64x32)
    const int row = c >> 2;                // d row
    const int off = (c & 3) * 8;           // key offset within row
    async_ld16(vlds + (unsigned)(row * 32 + off) * 2,
               Vh + (size_t)row * SEQ + kb + off);
  }
}

__global__ void __launch_bounds__(128)
flash_attn_kernel(const bf16* __restrict__ Q, const bf16* __restrict__ Km,
                  const bf16* __restrict__ Vt, bf16* __restrict__ AO) {
  __shared__ alignas(16) bf16 Kl[2][32 * HDIM];   // 2 x 4 KB
  __shared__ alignas(16) bf16 Vl[2][HDIM * 32];   // 2 x 4 KB
  __shared__ alignas(16) bf16 plds[4][16][32];    // per-wave P transpose

  const int tid  = threadIdx.x;
  const int lane = tid & 31;
  const int wave = tid >> 5;
  const int hi   = lane >> 4;
  const int ln   = lane & 15;

  const int h  = blockIdx.y;
  const int b  = blockIdx.z;
  const int bh = b * HEADS + h;
  const int q0 = blockIdx.x * 64 + wave * 16;

  const bf16* Qh = Q  + (size_t)bh * SEQ * HDIM;
  const bf16* Kh = Km + (size_t)bh * SEQ * HDIM;
  const bf16* Vh = Vt + (size_t)bh * HDIM * SEQ;

  const unsigned kaddr[2] = {lds_off(&Kl[0][0]), lds_off(&Kl[1][0])};
  const unsigned vaddr[2] = {lds_off(&Vl[0][0]), lds_off(&Vl[1][0])};

  // Q tile (16 rows x 64 d) lives in registers for the whole kernel.
  const v16bf qa0 = ld_a_bf16(Qh + (size_t)(q0 + ln) * HDIM + 0,  hi);
  const v16bf qa1 = ld_a_bf16(Qh + (size_t)(q0 + ln) * HDIM + 32, hi);

  // Ones B-operand: row-sum of P via the matrix core.
  v16bf onesb;
#pragma unroll
  for (int j = 0; j < 16; ++j) onesb[j] = (bf16)1.0f;

  v8f acc[4] = {};                        // O tile: 16 x 64 f32
  float mrun[8], lrun[8];
#pragma unroll
  for (int r = 0; r < 8; ++r) { mrun[r] = -__builtin_inff(); lrun[r] = 0.0f; }

  // softmax in base-2: p = exp2(s*sc2 - m2), sc2 = (1/sqrt(D)) * log2(e)
  const float sc2 = 0.125f * 1.44269504088896f;

  bf16* Lp = &plds[wave][0][0];

  // Preload first K/V tile.
  stage_kv(kaddr[0], vaddr[0], Kh, Vh, 0, tid);

  for (int kb = 0, it = 0; kb < SEQ; kb += 32, ++it) {
    const int cur = it & 1;

    // Own DMA for buffer `cur` done; barrier publishes it block-wide and
    // guarantees everyone is finished reading buffer `cur^1`.
    wait_asynccnt0();
    __syncthreads();

    // Kick DMA for the next tile into the other buffer; overlaps compute.
    if (kb + 32 < SEQ)
      stage_kv(kaddr[cur ^ 1], vaddr[cur ^ 1], Kh, Vh, kb + 32, tid);

    const bf16* Kc = &Kl[cur][0];
    const bf16* Vc = &Vl[cur][0];

    // ---- scores: two 16x16 f32 tiles (B operands from LDS) ---------------
    const bf16* krow0 = Kc + (size_t)ln * HDIM;          // keys kb..kb+15
    const bf16* krow1 = Kc + (size_t)(16 + ln) * HDIM;   // keys kb+16..kb+31
    v8f s0 = {}, s1 = {};
    s0 = wmma_bf16(qa0, ld_b_bf16(krow0 + 0,  hi), s0);
    s0 = wmma_bf16(qa1, ld_b_bf16(krow0 + 32, hi), s0);
    s1 = wmma_bf16(qa0, ld_b_bf16(krow1 + 0,  hi), s1);
    s1 = wmma_bf16(qa1, ld_b_bf16(krow1 + 32, hi), s1);

    // ---- online softmax: max via lane shuffles, exp, rescale -------------
    v8f p0, p1;
    float corr[8];
#pragma unroll
    for (int r = 0; r < 8; ++r) {
      const float a0 = s0[r] * sc2;
      const float a1 = s1[r] * sc2;
      float mx = fmaxf(a0, a1);           // row r+8*hi, reduce over 16 lanes
      mx = fmaxf(mx, __shfl_xor(mx, 1, 32));
      mx = fmaxf(mx, __shfl_xor(mx, 2, 32));
      mx = fmaxf(mx, __shfl_xor(mx, 4, 32));
      mx = fmaxf(mx, __shfl_xor(mx, 8, 32));
      const float mnew = fmaxf(mrun[r], mx);
      corr[r] = exp2f(mrun[r] - mnew);    // 0 on first iteration
      mrun[r] = mnew;
      p0[r] = exp2f(a0 - mnew);
      p1[r] = exp2f(a1 - mnew);
      acc[0][r] *= corr[r]; acc[1][r] *= corr[r];
      acc[2][r] *= corr[r]; acc[3][r] *= corr[r];
    }

    // ---- transpose P (C layout -> A layout) via this wave's LDS slice ----
#pragma unroll
    for (int r = 0; r < 8; ++r) {
      const int m = r + hi * 8;
      Lp[m * 32 + ln]      = (bf16)p0[r];
      Lp[m * 32 + 16 + ln] = (bf16)p1[r];
    }
    // same-wave DS RAW: LDS pipe is in-order; fence compiler + DScnt
    asm volatile("s_wait_dscnt 0" ::: "memory");
    const v16bf pa = ld_a_bf16(Lp + ln * 32, hi);   // P as 16x32 A operand

    // ---- row sums on the matrix core: rs = P * ones ----------------------
    v8f rsum = {};
    rsum = wmma_bf16(pa, onesb, rsum);
#pragma unroll
    for (int r = 0; r < 8; ++r) lrun[r] = lrun[r] * corr[r] + rsum[r];

    // ---- O += P * V (B operands from LDS) --------------------------------
#pragma unroll
    for (int nt = 0; nt < 4; ++nt) {
      const v16bf vb = ld_b_bf16(Vc + (size_t)(nt * 16 + ln) * 32, hi);
      acc[nt] = wmma_bf16(pa, vb, acc[nt]);
    }
  }

  // ---- normalize and store with heads re-merged: AO[b, s, h*64+d] --------
  float inv[8];
#pragma unroll
  for (int r = 0; r < 8; ++r) inv[r] = 1.0f / lrun[r];

#pragma unroll
  for (int nt = 0; nt < 4; ++nt) {
#pragma unroll
    for (int r = 0; r < 8; ++r) {
      const int s = q0 + r + hi * 8;
      const int d = h * HDIM + nt * 16 + ln;
      AO[((size_t)b * SEQ + s) * EMBED + d] = (bf16)(acc[nt][r] * inv[r]);
    }
  }
}

// ---------------------------------------------------------------------------
// Launch
// ---------------------------------------------------------------------------
extern "C" void kernel_launch(void* const* d_in, const int* in_sizes, int n_in,
                              void* d_out, int out_size, void* d_ws,
                              size_t ws_size, hipStream_t stream) {
  const float* query = (const float*)d_in[0];
  const float* key   = (const float*)d_in[1];
  const float* value = (const float*)d_in[2];
  const float* Wq    = (const float*)d_in[3];
  const float* bq    = (const float*)d_in[4];
  const float* Wk    = (const float*)d_in[5];
  const float* bk    = (const float*)d_in[6];
  const float* Wv    = (const float*)d_in[7];
  const float* bv    = (const float*)d_in[8];
  const float* Wo    = (const float*)d_in[9];
  const float* bo    = (const float*)d_in[10];

  const size_t elems = (size_t)BATCH * SEQ * EMBED;   // 4.19M
  bf16* Qb  = (bf16*)d_ws;          // [B,H,S,D] bf16
  bf16* Kb  = Qb  + elems;          // [B,H,S,D] bf16
  bf16* Vtb = Kb  + elems;          // [B,H,D,S] bf16
  bf16* AOb = Vtb + elems;          // [B,S,E]   bf16   (total 33.6 MB)

  const int M = BATCH * SEQ;        // 4096
  dim3 gg(M / 128, EMBED / 64);     // 32 x 16 blocks, 256 thr each

  gemm_wmma_kernel<false, 1><<<gg, 256, 0, stream>>>(query, Wq, bq, Qb,
                                                     M, EMBED, EMBED);
  gemm_wmma_kernel<false, 1><<<gg, 256, 0, stream>>>(key, Wk, bk, Kb,
                                                     M, EMBED, EMBED);
  gemm_wmma_kernel<false, 2><<<gg, 256, 0, stream>>>(value, Wv, bv, Vtb,
                                                     M, EMBED, EMBED);

  flash_attn_kernel<<<dim3(SEQ / 64, HEADS, BATCH), 128, 0, stream>>>(
      Qb, Kb, Vtb, AOb);

  gemm_wmma_kernel<true, 0><<<gg, 256, 0, stream>>>(AOb, Wo, bo, d_out,
                                                    M, EMBED, EMBED);
}
